// SelfAttention3D_24550033064608
// MI455X (gfx1250) — matrix operand order repaired and checked by reading, not compiled, not run on hardware
//
#include <hip/hip_runtime.h>

// ---------------------------------------------------------------------------
// CDNA5 (gfx1250) bf16-WMMA implementation of channel self-attention.
// Pipeline: cvt(f32->bf16) -> GEMM(qkv) -> scores(split-K WMMA, no LDS)
//           -> softmax+reduce -> attn*V (WMMA) -> GEMM(proj, +bias, f32 out)
// All matmuls use v_wmma_f32_16x16x32_bf16 (wave32, 16x16 tiles, K=32).
// GEMM inner loop: double-buffered LDS (1 barrier/K-step), register staging,
// bank-conflict-free padded LDS (row stride 40 shorts = 80B = 5x16B),
// global_prefetch of the tile two K-steps ahead.
// ---------------------------------------------------------------------------

typedef __attribute__((ext_vector_type(16))) __bf16 v16bf;
typedef __attribute__((ext_vector_type(8)))  float  v8f;

#define N_TOTAL 32768
#define C_CH    512
#define LDS_PAD 40   // shorts per LDS row: 80B, multiple of 16B, coprime-ish banks

union FragB {
  v16bf v;
  uint4 q[2];
};

__device__ __forceinline__ unsigned short f32_to_bf16_bits(float f) {
  union { float f; unsigned int u; } x; x.f = f;
  unsigned int r = x.u + 0x7FFFu + ((x.u >> 16) & 1u);  // round-to-nearest-even
  return (unsigned short)(r >> 16);
}

// --------------------------- f32 -> bf16 convert ---------------------------
__global__ void cvt_f32_bf16(const float* __restrict__ src,
                             unsigned short* __restrict__ dst, int n) {
  int i = (blockIdx.x * blockDim.x + threadIdx.x) * 4;
  if (i + 3 < n) {
    float4 f = *(const float4*)(src + i);
    ushort4 o;
    o.x = f32_to_bf16_bits(f.x);
    o.y = f32_to_bf16_bits(f.y);
    o.z = f32_to_bf16_bits(f.z);
    o.w = f32_to_bf16_bits(f.w);
    *(ushort4*)(dst + i) = o;
  } else {
    for (int j = i; j < n; ++j) dst[j] = f32_to_bf16_bits(src[j]);
  }
}

// --------------------------- generic WMMA GEMM -----------------------------
// C[b] = A[M x K] * B[b][K x N] (+bias). A shared across batch.
// Block: 256 thr (8 waves). Tile: 128(M) x 64(N), K-step 32, double-buffered.
template <bool OUT_BF16>
__global__ __launch_bounds__(256)
void gemm_wmma(const unsigned short* __restrict__ A,
               const unsigned short* __restrict__ B,
               void* __restrict__ C,
               const float* __restrict__ bias,
               int M, int K, int N,
               long long strideB_batch, long long strideC_batch) {
  __shared__ __align__(16) unsigned short As[2][128][LDS_PAD];  // [m][k]
  __shared__ __align__(16) unsigned short Bs[2][64][LDS_PAD];   // [n][k] (transposed)

  const int t    = threadIdx.x;
  const int wave = t >> 5;
  const int lane = t & 31;
  const int half = lane >> 4;
  const int l16  = lane & 15;
  const int m0   = blockIdx.y * 128;
  const int n0   = blockIdx.x * 64;
  const int b    = blockIdx.z;
  const unsigned short* Bb = B + (size_t)b * strideB_batch;

  // staging coordinates (per thread)
  const int mA = t >> 1, kA = (t & 1) * 16;  // A: 128 rows x 32 k, 32B/thread
  const int kB = t >> 3, nB = (t & 7) * 8;   // B: 32 k x 64 n, 16B/thread
  const unsigned short* aSrc = A + (size_t)(m0 + mA) * K + kA;
  const unsigned short* bSrc = Bb + (size_t)kB * N + n0 + nB;

  v8f acc[4] = {};
  const int kSteps = K >> 5;

  // prologue: stage tile 0 into buffer 0
  uint4 ra0 = *(const uint4*)(aSrc);
  uint4 ra1 = *(const uint4*)(aSrc + 8);
  uint4 rb  = *(const uint4*)(bSrc);
  {
    uint4* dA = (uint4*)&As[0][mA][kA];
    dA[0] = ra0;
    dA[1] = ra1;
    const unsigned short* ds = (const unsigned short*)&rb;
#pragma unroll
    for (int jj = 0; jj < 8; ++jj) {  // lane-rotated to spread LDS banks
      int j = (jj + lane) & 7;
      Bs[0][nB + j][kB] = ds[j];
    }
  }
  __syncthreads();

  for (int ks = 0; ks < kSteps; ++ks) {
    const int cur = ks & 1;
    const bool more = (ks + 1) < kSteps;
    if (more) {  // issue next-tile global loads before compute (latency overlap)
      ra0 = *(const uint4*)(aSrc + (ks + 1) * 32);
      ra1 = *(const uint4*)(aSrc + (ks + 1) * 32 + 8);
      rb  = *(const uint4*)(bSrc + (size_t)(ks + 1) * 32 * N);
      if (ks + 2 < kSteps) {  // pull tile ks+2 toward cache
        __builtin_prefetch(aSrc + (ks + 2) * 32, 0, 1);
        __builtin_prefetch(bSrc + (size_t)(ks + 2) * 32 * N, 0, 1);
      }
    }

    // compute from current buffer
    FragB a;  // 16x32 A fragment (ISA 16-bit A layout)
    const int m = wave * 16 + l16;
    a.q[0] = *(const uint4*)&As[cur][m][8 * half];
    a.q[1] = *(const uint4*)&As[cur][m][16 + 8 * half];
#pragma unroll
    for (int nt = 0; nt < 4; ++nt) {
      FragB bf;  // 32x16 B fragment: lane=col, table K order
      const int col = nt * 16 + l16;
      bf.q[0] = *(const uint4*)&Bs[cur][col][8 * half];
      bf.q[1] = *(const uint4*)&Bs[cur][col][16 + 8 * half];
      acc[nt] = __builtin_amdgcn_wmma_f32_16x16x32_bf16(
          false, a.v, false, bf.v, (short)0, acc[nt], false, false);
    }

    if (more) {  // stage next tile into alternate buffer
      uint4* dA = (uint4*)&As[1 - cur][mA][kA];
      dA[0] = ra0;
      dA[1] = ra1;
      const unsigned short* ds = (const unsigned short*)&rb;
#pragma unroll
      for (int jj = 0; jj < 8; ++jj) {
        int j = (jj + lane) & 7;
        Bs[1 - cur][nB + j][kB] = ds[j];
      }
    }
    __syncthreads();
  }

#pragma unroll
  for (int nt = 0; nt < 4; ++nt) {
#pragma unroll
    for (int r = 0; r < 8; ++r) {
      const int row = m0 + wave * 16 + r + 8 * half;
      const int col = n0 + nt * 16 + l16;
      float v = acc[nt][r];
      if (bias) v += bias[row];
      if (OUT_BF16)
        ((unsigned short*)C)[(size_t)b * strideC_batch + (size_t)row * N + col] =
            f32_to_bf16_bits(v);
      else
        ((float*)C)[(size_t)b * strideC_batch + (size_t)row * N + col] = v;
    }
  }
}

// ------------------- scores: S_part[chunk][bh] = Q * K^T -------------------
// Both operands stream contiguously along n -> direct global b128 loads.
// Grid: (16 chunks of n=2048, 16 bh). Wave w: m-subtile w&3, n-subtiles 2*(w>>2)+{0,1}.
__global__ __launch_bounds__(256)
void scores_wmma(const unsigned short* __restrict__ qkv,
                 float* __restrict__ Spart) {
  const int t = threadIdx.x, wave = t >> 5, lane = t & 31;
  const int half = lane >> 4, l16 = lane & 15;
  const int chunk = blockIdx.x;
  const int bh = blockIdx.y;
  const int b = bh >> 3, h = bh & 7;
  const int mt = wave & 3;
  const int ntBase = (wave >> 2) * 2;
  const size_t qbase = ((size_t)b * 1536 + (size_t)h * 64) * N_TOTAL;
  const size_t kbase = ((size_t)b * 1536 + 512 + (size_t)h * 64) * N_TOTAL;

  v8f acc[2] = {};
  const int nStart = chunk * 2048;
  for (int kk = 0; kk < 2048; kk += 32) {
    const int np = nStart + kk;
    FragB a;
    const unsigned short* ap = qkv + qbase + (size_t)(mt * 16 + l16) * N_TOTAL + np;
    a.q[0] = *(const uint4*)(ap + 8 * half);
    a.q[1] = *(const uint4*)(ap + 16 + 8 * half);
#pragma unroll
    for (int i = 0; i < 2; ++i) {
      const int nt = ntBase + i;
      FragB bf;  // B[kk][j] = Kmat[j][n] -> lane=col j, contiguous along n
      const unsigned short* bp = qkv + kbase + (size_t)(nt * 16 + l16) * N_TOTAL + np;
      bf.q[0] = *(const uint4*)(bp + 8 * half);
      bf.q[1] = *(const uint4*)(bp + 16 + 8 * half);
      acc[i] = __builtin_amdgcn_wmma_f32_16x16x32_bf16(
          false, a.v, false, bf.v, (short)0, acc[i], false, false);
    }
  }
  float* out = Spart + ((size_t)chunk * 16 + bh) * 64 * 64;
#pragma unroll
  for (int i = 0; i < 2; ++i) {
    const int nt = ntBase + i;
#pragma unroll
    for (int r = 0; r < 8; ++r) {
      const int m = mt * 16 + r + 8 * half;
      const int n = nt * 16 + l16;
      out[m * 64 + n] = acc[i][r];
    }
  }
}

// ---------------- softmax (reduces 16 split-K partials) --------------------
__global__ void softmax_rows(const float* __restrict__ Spart,
                             unsigned short* __restrict__ attnb) {
  const int bh = blockIdx.x;
  const int i = threadIdx.x;  // row 0..63
  float row[64];
#pragma unroll 4
  for (int j = 0; j < 64; ++j) {
    float s = 0.f;
    for (int c = 0; c < 16; ++c)
      s += Spart[(((size_t)c * 16 + bh) * 64 + i) * 64 + j];
    row[j] = s * 0.125f;  // DIM_HEAD^-0.5
  }
  float mx = row[0];
  for (int j = 1; j < 64; ++j) mx = fmaxf(mx, row[j]);
  float sum = 0.f;
  for (int j = 0; j < 64; ++j) { row[j] = __expf(row[j] - mx); sum += row[j]; }
  const float inv = 1.f / sum;
  for (int j = 0; j < 64; ++j)
    attnb[((size_t)bh * 64 + i) * 64 + j] = f32_to_bf16_bits(row[j] * inv);
}

// --------------------------- out = attn * V --------------------------------
// Block tile: 128 channel rows (2 heads) x 64 n. V tile staged transposed.
__global__ __launch_bounds__(256)
void attnv_wmma(const unsigned short* __restrict__ qkv,
                const unsigned short* __restrict__ attnb,
                unsigned short* __restrict__ outb) {
  __shared__ __align__(16) unsigned short Vs[64][136];  // [n_local][krow], 272B stride
  const int t = threadIdx.x, wave = t >> 5, lane = t & 31;
  const int half = lane >> 4, l16 = lane & 15;
  const int n0 = blockIdx.x * 64;
  const int mt2 = blockIdx.y;  // channel rows mt2*128..+128 (heads 2*mt2, 2*mt2+1)
  const int b = blockIdx.z;

  {  // stage V rows (2 heads = 128 k-rows) x 64 n, transposed
    const int krow = t >> 1;
    const int n_l0 = (t & 1) * 32;
    const unsigned short* src =
        qkv + ((size_t)b * 1536 + 1024 + mt2 * 128 + krow) * N_TOTAL + n0 + n_l0;
#pragma unroll
    for (int g = 0; g < 4; ++g) {
      uint4 d = *(const uint4*)(src + g * 8);
      const unsigned short* ds = (const unsigned short*)&d;
#pragma unroll
      for (int jj = 0; jj < 8; ++jj) {
        int j = (jj + lane) & 7;
        Vs[n_l0 + g * 8 + j][krow] = ds[j];
      }
    }
  }
  __syncthreads();

  const int hLocal = wave >> 2;  // 0 or 1
  const int h = mt2 * 2 + hLocal;
  const int mHead = (wave & 3) * 16;  // wave's row block within its head
  const unsigned short* attnRow =
      attnb + (((size_t)(b * 8 + h) * 64) + mHead + l16) * 64;

  v8f acc[4] = {};
#pragma unroll
  for (int kk = 0; kk < 64; kk += 32) {
    FragB a;
    a.q[0] = *(const uint4*)(attnRow + kk + 8 * half);
    a.q[1] = *(const uint4*)(attnRow + kk + 16 + 8 * half);
    const int krow0 = hLocal * 64 + kk;
#pragma unroll
    for (int nt = 0; nt < 4; ++nt) {
      FragB bf;
      const int col = nt * 16 + l16;
      bf.q[0] = *(const uint4*)&Vs[col][krow0 + 8 * half];
      bf.q[1] = *(const uint4*)&Vs[col][krow0 + 16 + 8 * half];
      acc[nt] = __builtin_amdgcn_wmma_f32_16x16x32_bf16(
          false, a.v, false, bf.v, (short)0, acc[nt], false, false);
    }
  }

  const int rowBase = mt2 * 128 + wave * 16;
#pragma unroll
  for (int nt = 0; nt < 4; ++nt) {
#pragma unroll
    for (int r = 0; r < 8; ++r) {
      const int row = rowBase + r + 8 * half;
      const int col = n0 + nt * 16 + l16;
      outb[((size_t)b * C_CH + row) * N_TOTAL + col] = f32_to_bf16_bits(acc[nt][r]);
    }
  }
}

// ------------------------------- launcher ----------------------------------
extern "C" void kernel_launch(void* const* d_in, const int* in_sizes, int n_in,
                              void* d_out, int out_size, void* d_ws, size_t ws_size,
                              hipStream_t stream) {
  const float* x      = (const float*)d_in[0];  // (2,512,32,32,32)
  const float* w_qkv  = (const float*)d_in[1];  // (1536,512)
  const float* w_proj = (const float*)d_in[2];  // (512,512)
  const float* b_proj = (const float*)d_in[3];  // (512,)

  const size_t nX    = 2ull * 512 * N_TOTAL;   // 33,554,432
  const size_t nWqkv = 1536ull * 512;          // 786,432
  const size_t nWprj = 512ull * 512;           // 262,144
  const size_t nQkv  = 2ull * 1536 * N_TOTAL;  // 100,663,296

  // workspace layout (bytes); x_bf16 region is reused for attn output
  char* ws = (char*)d_ws;
  unsigned short* xb     = (unsigned short*)(ws);                      // 64 MiB (also outb)
  unsigned short* wqkvb  = (unsigned short*)(ws + nX * 2);
  unsigned short* wprjb  = (unsigned short*)(ws + nX * 2 + nWqkv * 2);
  unsigned short* qkvb   = (unsigned short*)(ws + nX * 2 + nWqkv * 2 + nWprj * 2);
  float*          Spart  = (float*)(ws + nX * 2 + nWqkv * 2 + nWprj * 2 + nQkv * 2);
  unsigned short* attnb  = (unsigned short*)((char*)Spart + 16ull * 16 * 64 * 64 * 4);
  unsigned short* outb   = xb;  // reuse: x dead after qkv GEMM

  // 1) convert inputs to bf16
  cvt_f32_bf16<<<dim3((unsigned)((nX / 4 + 255) / 256)), 256, 0, stream>>>(x, xb, (int)nX);
  cvt_f32_bf16<<<dim3((unsigned)((nWqkv / 4 + 255) / 256)), 256, 0, stream>>>(w_qkv, wqkvb, (int)nWqkv);
  cvt_f32_bf16<<<dim3((unsigned)((nWprj / 4 + 255) / 256)), 256, 0, stream>>>(w_proj, wprjb, (int)nWprj);

  // 2) qkv = w_qkv * x : [1536x512]*[512x32768] per batch
  gemm_wmma<true><<<dim3(N_TOTAL / 64, 1536 / 128, 2), 256, 0, stream>>>(
      wqkvb, xb, qkvb, nullptr, 1536, 512, N_TOTAL,
      (long long)512 * N_TOTAL, (long long)1536 * N_TOTAL);

  // 3) split-K attention scores (16 chunks x 16 bh)
  scores_wmma<<<dim3(16, 16), 256, 0, stream>>>(qkvb, Spart);

  // 4) reduce partials + softmax -> bf16 attn
  softmax_rows<<<dim3(16), 64, 0, stream>>>(Spart, attnb);

  // 5) out = attn * V
  attnv_wmma<<<dim3(N_TOTAL / 64, 4, 2), 256, 0, stream>>>(qkvb, attnb, outb);

  // 6) final projection (f32 out, +bias)
  gemm_wmma<false><<<dim3(N_TOTAL / 64, 512 / 128, 2), 256, 0, stream>>>(
      wprjb, outb, d_out, b_proj, 512, 512, N_TOTAL,
      (long long)512 * N_TOTAL, (long long)512 * N_TOTAL);
}
